// VGAEModel_27049704030766
// MI455X (gfx1250) — matrix operand order repaired and checked by reading, not compiled, not run on hardware
//
#include <hip/hip_runtime.h>
#include <hip/hip_bf16.h>

// ---------------------------------------------------------------------------
// VGAE forward for MI455X (gfx1250, wave32, WMMA bf16)
// ---------------------------------------------------------------------------

typedef __attribute__((ext_vector_type(16))) __bf16 v16bf;
typedef __attribute__((ext_vector_type(8)))  __bf16 v8bf;
typedef __attribute__((ext_vector_type(8)))  float  v8f;

#define NNODES 20000
#define NEDGES 640000
#define INC    256
#define HID    64
#define MAXLS  10.0f

// ------------------------------ helpers ------------------------------------

__global__ void k_fill_f32(float* __restrict__ p, float v, int n) {
    int i = blockIdx.x * blockDim.x + threadIdx.x;
    if (i < n) p[i] = v;
}

__global__ void k_fill_i32(int* __restrict__ p, int v, int n) {
    int i = blockIdx.x * blockDim.x + threadIdx.x;
    if (i < n) p[i] = v;
}

__global__ void k_f32_to_bf16(const float* __restrict__ s, __bf16* __restrict__ d, int n) {
    int i = blockIdx.x * blockDim.x + threadIdx.x;
    if (i < n) d[i] = (__bf16)s[i];
}

// transpose-convert W[K][N] (row major fp32) -> Wt[N][K] (bf16), so B-matrix
// WMMA fragments become contiguous 32B loads per lane.
__global__ void k_transpose_bf16(const float* __restrict__ s, __bf16* __restrict__ d,
                                 int K, int N) {
    int i = blockIdx.x * blockDim.x + threadIdx.x;
    if (i < K * N) {
        int k = i / N, n = i % N;
        d[n * K + k] = (__bf16)s[i];
    }
}

// ------------------------- GCN normalization -------------------------------

__global__ void k_deg(const long long* __restrict__ dst, int* __restrict__ deg, int E) {
    int e = blockIdx.x * blockDim.x + threadIdx.x;
    if (e < E) atomicAdd(&deg[(int)dst[e]], 1);
}

__global__ void k_dinv(const int* __restrict__ deg, float* __restrict__ dinv, int n) {
    int i = blockIdx.x * blockDim.x + threadIdx.x;
    if (i < n) dinv[i] = rsqrtf((float)deg[i]);   // deg >= 1 (self-loop)
}

// one wave per edge (incl. N self-loops appended past E); fp32 global atomics
__global__ void k_agg(const long long* __restrict__ ei, const float* __restrict__ dinv,
                      const float* __restrict__ feat, float* __restrict__ out,
                      int E, int n, int F) {
    int gw   = (blockIdx.x * blockDim.x + threadIdx.x) >> 5;
    int lane = threadIdx.x & 31;
    if (gw >= E + n) return;
    int s, d;
    if (gw < E) { s = (int)ei[gw]; d = (int)ei[E + gw]; }
    else        { s = d = gw - E; }
    float w = dinv[s] * dinv[d];
    const float* fs = feat + (size_t)s * F;
    float*       od = out  + (size_t)d * F;
    for (int f = lane; f < F; f += 32)
        atomicAdd(od + f, fs[f] * w);
}

__global__ void k_bias_relu_bf16(const float* __restrict__ agg, const float* __restrict__ b,
                                 __bf16* __restrict__ hb, int n) {
    int i = blockIdx.x * blockDim.x + threadIdx.x;
    if (i < n) hb[i] = (__bf16)fmaxf(agg[i] + b[i & (HID - 1)], 0.0f);
}

// ------------------------------- GEMMs -------------------------------------
// Wave-level 16x16 output tile; A lane(m=lane&15): K = (lane>>4)*16 + {0..15};
// B lane(n=lane&15): K = (lane>>4)*16 + {0..15};
// D VGPR r -> row (lane>>4)*8 + r, col = lane&15.   (D per ISA 7.12.2; A/B k order is this code's own)

__device__ __forceinline__ v16bf load_a_frag(const __bf16* __restrict__ row,
                                             int k0, int half) {
    return *(const v16bf*)(row + k0 + half * 16);
}

// h0 = x @ W1   [20000x256]x[256x64]; block = 8 waves = 2 row-tiles x 4 col-tiles
__global__ void k_gemm_x_w1(const __bf16* __restrict__ xb,
                            const __bf16* __restrict__ w1t,
                            float* __restrict__ h0) {
    const int wave = threadIdx.x >> 5, lane = threadIdx.x & 31;
    const int rt = blockIdx.x * 2 + (wave & 1);   // 0..1249
    const int ct = wave >> 1;                     // 0..3
    const int m = lane & 15, half = lane >> 4;
    const __bf16* arow = xb  + (size_t)(rt * 16 + m) * INC;
    const __bf16* bcol = w1t + (size_t)(ct * 16 + m) * INC;
    v8f c = {};
#pragma unroll
    for (int kk = 0; kk < INC / 32; ++kk) {
        const int k0 = kk * 32;
        v16bf a = load_a_frag(arow, k0, half);
        v16bf b = *(const v16bf*)(bcol + k0 + half * 16);
        c = __builtin_amdgcn_wmma_f32_16x16x32_bf16(false, a, false, b,
                                                    (short)0, c, false, false);
    }
    float* o = h0 + (size_t)(rt * 16 + half * 8) * HID + ct * 16 + m;
#pragma unroll
    for (int r = 0; r < 8; ++r) o[(size_t)r * HID] = c[r];
}

// t2 = h @ [Wmu | Wls]  -> [20000 x 128]; block = 1 row-tile x 8 col-tiles
__global__ void k_gemm_h(const __bf16* __restrict__ hb,
                         const __bf16* __restrict__ wmut,
                         const __bf16* __restrict__ wlst,
                         float* __restrict__ t2) {
    const int wave = threadIdx.x >> 5, lane = threadIdx.x & 31;
    const int rt = blockIdx.x;
    const int m = lane & 15, half = lane >> 4;
    const __bf16* wt = (wave < 4) ? wmut : wlst;
    const int ctn = wave & 3;
    const __bf16* arow = hb + (size_t)(rt * 16 + m) * HID;
    const __bf16* bcol = wt + (size_t)(ctn * 16 + m) * HID;
    v8f c = {};
#pragma unroll
    for (int kk = 0; kk < 2; ++kk) {
        const int k0 = kk * 32;
        v16bf a = load_a_frag(arow, k0, half);
        v16bf b = *(const v16bf*)(bcol + k0 + half * 16);
        c = __builtin_amdgcn_wmma_f32_16x16x32_bf16(false, a, false, b,
                                                    (short)0, c, false, false);
    }
    float* o = t2 + (size_t)(rt * 16 + half * 8) * 128 + wave * 16 + m;
#pragma unroll
    for (int r = 0; r < 8; ++r) o[(size_t)r * 128] = c[r];
}

// z = mu + eps * exp(min(logstd, 10)) -> bf16
__global__ void k_z(const float* __restrict__ agg2, const float* __restrict__ bmu,
                    const float* __restrict__ bls, const float* __restrict__ eps,
                    __bf16* __restrict__ zb, int n) {
    int i = blockIdx.x * blockDim.x + threadIdx.x;
    if (i >= n) return;
    int node = i >> 6, f = i & 63;
    float mu = agg2[(size_t)node * 128 + f] + bmu[f];
    float ls = fminf(agg2[(size_t)node * 128 + 64 + f] + bls[f], MAXLS);
    zb[i] = (__bf16)(mu + eps[i] * __expf(ls));
}

// adj = sigmoid(z @ z^T): wave = one 16x16 tile, 2 bf16 WMMAs (K=64).
// z (2.5MB bf16) lives in L2; output stream uses non-temporal stores.
__global__ void k_decoder(const __bf16* __restrict__ zb, float* __restrict__ out,
                          int n) {
    const int wave = threadIdx.x >> 5, lane = threadIdx.x & 31;
    const int jt = blockIdx.x * 8 + wave;       // column tile
    if (jt * 16 >= n) return;                   // wave-uniform: EXEC stays all-1
    const int it = blockIdx.y;                  // row tile
    const int m = lane & 15, half = lane >> 4;
    const __bf16* arow = zb + (size_t)(it * 16 + m) * HID;
    const __bf16* brow = zb + (size_t)(jt * 16 + m) * HID;  // B col n == z row j
    v8f c = {};
#pragma unroll
    for (int kk = 0; kk < 2; ++kk) {
        const int k0 = kk * 32;
        v16bf a = load_a_frag(arow, k0, half);
        v16bf b = *(const v16bf*)(brow + k0 + half * 16);
        c = __builtin_amdgcn_wmma_f32_16x16x32_bf16(false, a, false, b,
                                                    (short)0, c, false, false);
    }
    const size_t col = (size_t)jt * 16 + m;
    const size_t rowbase = (size_t)it * 16 + half * 8;
#pragma unroll
    for (int r = 0; r < 8; ++r) {
        float v = 1.0f / (1.0f + __expf(-c[r]));
        __builtin_nontemporal_store(v, out + (rowbase + r) * (size_t)n + col);
    }
}

// ------------------------------ launcher -----------------------------------

extern "C" void kernel_launch(void* const* d_in, const int* in_sizes, int n_in,
                              void* d_out, int out_size, void* d_ws, size_t ws_size,
                              hipStream_t stream) {
    const float*     x   = (const float*)d_in[0];
    const long long* ei  = (const long long*)d_in[1];   // int64 (2, E)
    const float*     W1  = (const float*)d_in[2];
    const float*     b1  = (const float*)d_in[3];
    const float*     Wmu = (const float*)d_in[4];
    const float*     bmu = (const float*)d_in[5];
    const float*     Wls = (const float*)d_in[6];
    const float*     bls = (const float*)d_in[7];
    const float*     eps = (const float*)d_in[8];
    float* adj = (float*)d_out;

    const int N = NNODES, E = NEDGES;

    // workspace carve-out (256B aligned slices); ~33 MB with aliasing
    char* ws = (char*)d_ws;
    size_t off = 0;
    auto carve = [&](size_t bytes) -> char* {
        char* p = ws + off;
        off = (off + bytes + 255) & ~(size_t)255;
        return p;
    };
    int*    deg  = (int*)carve((size_t)N * 4);
    float*  dinv = (float*)carve((size_t)N * 4);
    __bf16* xb   = (__bf16*)carve((size_t)N * INC * 2);   // 10.24 MB (aliased by t2)
    __bf16* w1t  = (__bf16*)carve((size_t)HID * INC * 2);
    __bf16* wmut = (__bf16*)carve((size_t)HID * HID * 2);
    __bf16* wlst = (__bf16*)carve((size_t)HID * HID * 2);
    float*  h0   = (float*)carve((size_t)N * HID * 4);    // 5.12 MB (aliased by zb)
    float*  agg1 = (float*)carve((size_t)N * HID * 4);
    __bf16* hb   = (__bf16*)carve((size_t)N * HID * 2);
    float*  agg2 = (float*)carve((size_t)N * 128 * 4);
    float*  t2 = (float*)xb;      // xb dead after GEMM1
    __bf16* zb = (__bf16*)h0;     // h0 dead after first aggregation

    const int T = 256;
    auto blks = [](long long work, int per) { return (unsigned)((work + per - 1) / per); };

    // 1) degrees (self-loop via init=1) and D^{-1/2}
    k_fill_i32<<<blks(N, T), T, 0, stream>>>(deg, 1, N);
    k_deg<<<blks(E, T), T, 0, stream>>>(ei + E, deg, E);   // dst row
    k_dinv<<<blks(N, T), T, 0, stream>>>(deg, dinv, N);

    // 2) precision prep: x -> bf16, weights -> transposed bf16
    k_f32_to_bf16<<<blks((long long)N * INC, T), T, 0, stream>>>(x, xb, N * INC);
    k_transpose_bf16<<<blks(INC * HID, T), T, 0, stream>>>(W1, w1t, INC, HID);
    k_transpose_bf16<<<blks(HID * HID, T), T, 0, stream>>>(Wmu, wmut, HID, HID);
    k_transpose_bf16<<<blks(HID * HID, T), T, 0, stream>>>(Wls, wlst, HID, HID);

    // 3) h0 = x @ W1 (WMMA bf16)
    k_gemm_x_w1<<<N / 32, T, 0, stream>>>(xb, w1t, h0);

    // 4) aggregate conv1:  agg1[dst] += norm * h0[src]
    k_fill_f32<<<blks((long long)N * HID, T), T, 0, stream>>>(agg1, 0.0f, N * HID);
    k_agg<<<blks((long long)(E + N) * 32, T), T, 0, stream>>>(ei, dinv, h0, agg1, E, N, HID);

    // 5) h = relu(agg1 + b1) -> bf16
    k_bias_relu_bf16<<<blks((long long)N * HID, T), T, 0, stream>>>(agg1, b1, hb, N * HID);

    // 6) t2 = h @ [Wmu | Wls] (WMMA bf16, fused)
    k_gemm_h<<<N / 16, T, 0, stream>>>(hb, wmut, wlst, t2);

    // 7) aggregate conv2+conv3 together (128 features per edge)
    k_fill_f32<<<blks((long long)N * 128, T), T, 0, stream>>>(agg2, 0.0f, N * 128);
    k_agg<<<blks((long long)(E + N) * 32, T), T, 0, stream>>>(ei, dinv, t2, agg2, E, N, 128);

    // 8) reparameterize -> z (bf16, cached in L2 for the decoder)
    k_z<<<blks((long long)N * HID, T), T, 0, stream>>>(agg2, bmu, bls, eps, zb, N * HID);

    // 9) adj = sigmoid(z z^T): 1.6 GB NT-store stream, WMMA bf16 compute
    dim3 grid((N / 16 + 7) / 8, N / 16);
    k_decoder<<<grid, T, 0, stream>>>(zb, adj, N);
}
